// ComplexPointNetwork_71244917506299
// MI455X (gfx1250) — matrix operand order repaired and checked by reading, compiled, not run on hardware
//
#include <hip/hip_runtime.h>
#include <hip/hip_bf16.h>
#include <stdint.h>

// ---------------------------------------------------------------------------
// Types for gfx1250 WMMA (wave32): v_wmma_f32_16x16x32_bf16
// ---------------------------------------------------------------------------
typedef __bf16 bf16;
typedef __attribute__((ext_vector_type(16))) __bf16 v16bf;
typedef __attribute__((ext_vector_type(8)))  float  v8f;

union FragU { uint4 u[2]; v16bf v; };

// A-matrix fragment (16x32 bf16, MxK). lane: m = lane&15, g = lane>>4.
// VGPR0-3 hold K = k0 + 8g + {0..7}; VGPR4-7 hold K = k0 + 8g + 16 + {0..7}.
__device__ inline v16bf load_a_frag(const bf16* __restrict__ X, int row, int ld,
                                    int k0, int g) {
    const bf16* p = X + (size_t)row * ld + k0 + g * 8;
    FragU f;
    f.u[0] = *reinterpret_cast<const uint4*>(p);
    f.u[1] = *reinterpret_cast<const uint4*>(p + 16);
    return f.v;
}

// B-matrix fragment (32x16 bf16, KxN) from row-major W[n][k]: B[k][n]=W[col0+n][k].
// lane: n = lane&15, g = lane>>4. VGPR0-3 hold K = k0 + 8g + {0..7}; VGPR4-7 hold K = k0 + 8g + 16 + {0..7}.
__device__ inline v16bf load_b_frag(const bf16* __restrict__ W, int wrow, int ld,
                                    int k0, int g) {
    const bf16* p = W + (size_t)wrow * ld + k0 + g * 8;
    FragU f;
    f.u[0] = *reinterpret_cast<const uint4*>(p);
    f.u[1] = *reinterpret_cast<const uint4*>(p + 16);
    return f.v;
}

// ---------------------------------------------------------------------------
// GEMM + fused BN column statistics.
// H(nrows x CT*16, f32) = X(nrows x KF*32, bf16) @ W(CT*16 x KF*32)^T
// One wave per 16-row stripe. A frags register-resident across the column
// sweep; B frags double-buffered so weight-load latency hides behind WMMAs.
// Column sum / sumsq reduced through LDS, one global atomic per column/block.
// Requires nrows % (16 * blockDim/32 * gridDim) == 0 (exact: 120000 = 16*4*1875).
// ---------------------------------------------------------------------------
template <int KF, int CT>
__global__ __launch_bounds__(128) void gemm_bf16_kernel(
    const bf16* __restrict__ X, const bf16* __restrict__ W,
    float* __restrict__ H, float* __restrict__ gsum, float* __restrict__ gsq,
    int nrows) {
    constexpr int K    = KF * 32;
    constexpr int outW = CT * 16;

    __shared__ float lsum[outW];
    __shared__ float lsq[outW];
    for (int i = threadIdx.x; i < outW; i += 128) { lsum[i] = 0.f; lsq[i] = 0.f; }
    __syncthreads();

    const int lane = threadIdx.x & 31;
    const int n    = lane & 15;
    const int g    = lane >> 4;
    const int wid  = blockIdx.x * 4 + (threadIdx.x >> 5);
    const int row0 = wid * 16;

    v16bf afrag[KF];
#pragma unroll
    for (int f = 0; f < KF; ++f)
        afrag[f] = load_a_frag(X, row0 + n, K, f * 32, g);

    v16bf bfr[2][KF];
#pragma unroll
    for (int f = 0; f < KF; ++f)
        bfr[0][f] = load_b_frag(W, n, K, f * 32, g);

#pragma unroll
    for (int c = 0; c < CT; ++c) {
        const int cur = c & 1;
        if (c + 1 < CT) {
#pragma unroll
            for (int f = 0; f < KF; ++f)
                bfr[cur ^ 1][f] = load_b_frag(W, (c + 1) * 16 + n, K, f * 32, g);
        }
        v8f acc = {};
#pragma unroll
        for (int f = 0; f < KF; ++f)
            acc = __builtin_amdgcn_wmma_f32_16x16x32_bf16(
                false, afrag[f], false, bfr[cur][f], (short)0, acc, false, false);

        const int col0 = c * 16;
        // D layout: column = col0 + n, rows m = 8g + r
        float* out = H + (size_t)(row0 + g * 8) * outW + col0 + n;
        float s = 0.f, q = 0.f;
#pragma unroll
        for (int r = 0; r < 8; ++r) {
            float v = acc[r];
            out[(size_t)r * outW] = v;
            s += v; q += v * v;
        }
        atomicAdd(&lsum[col0 + n], s);
        atomicAdd(&lsq[col0 + n], q);
    }

    __syncthreads();
    for (int i = threadIdx.x; i < outW; i += 128) {
        atomicAdd(&gsum[i], lsum[i]);
        atomicAdd(&gsq[i], lsq[i]);
    }
}

// ---------------------------------------------------------------------------
// Layer 1: H1 = feat @ w1^T (C_IN=5) with fused column stats.
// Thread owns one column, strides over rows; stats in registers, one global
// atomic pair per (thread, launch).  grid: 512 blocks x 256 thr (2 rows/iter).
// ---------------------------------------------------------------------------
__global__ __launch_bounds__(256) void layer1_kernel(
    const float* __restrict__ feat, const float* __restrict__ w1,
    float* __restrict__ H, float* __restrict__ gsum, float* __restrict__ gsq,
    int nrows) {
    __shared__ float sw[128 * 5];
    for (int i = threadIdx.x; i < 640; i += 256) sw[i] = w1[i];
    __syncthreads();
    const int col = threadIdx.x & 127;
    const int rl  = threadIdx.x >> 7;   // 0..1
    const float* w = sw + col * 5;
    float s = 0.f, q = 0.f;
    for (int r = blockIdx.x * 2 + rl; r < nrows; r += gridDim.x * 2) {
        const float* f = feat + (size_t)r * 5;
        float acc = f[0]*w[0] + f[1]*w[1] + f[2]*w[2] + f[3]*w[3] + f[4]*w[4];
        H[(size_t)r * 128 + col] = acc;
        s += acc; q += acc * acc;
    }
    atomicAdd(&gsum[col], s);
    atomicAdd(&gsq[col], q);
}

// ---------------------------------------------------------------------------
__global__ void zero_stats_kernel(float* sum, float* sumsq, int width) {
    int i = blockIdx.x * blockDim.x + threadIdx.x;
    if (i < width) { sum[i] = 0.f; sumsq[i] = 0.f; }
}

__global__ void finalize_kernel(const float* __restrict__ sum,
                                const float* __restrict__ sumsq,
                                const float* __restrict__ gamma,
                                const float* __restrict__ beta,
                                float* __restrict__ scale,
                                float* __restrict__ bias, int width, float invN) {
    int i = blockIdx.x * blockDim.x + threadIdx.x;
    if (i >= width) return;
    float m = sum[i] * invN;
    float v = fmaxf(sumsq[i] * invN - m * m, 0.f);
    float s = gamma[i] * rsqrtf(v + 1e-5f);
    scale[i] = s;
    bias[i]  = beta[i] - m * s;
}

// ---------------------------------------------------------------------------
// normalize: x = (maybe relu)( H*scale + bias (+ resid) ); emit bf16 (+fp32)
// ---------------------------------------------------------------------------
__global__ __launch_bounds__(256) void normalize_kernel(
    const float* __restrict__ H, const float* __restrict__ scale,
    const float* __restrict__ bias, const float* __restrict__ resid,
    float* __restrict__ out_f32, bf16* __restrict__ out_bf,
    int total, int width, int do_relu) {
    int i4 = blockIdx.x * blockDim.x + threadIdx.x;
    size_t idx = (size_t)i4 * 4;
    if (idx >= (size_t)total) return;
    int c = (int)(idx % width);
    float4 h  = *reinterpret_cast<const float4*>(H + idx);
    float4 sc = *reinterpret_cast<const float4*>(scale + c);
    float4 bi = *reinterpret_cast<const float4*>(bias + c);
    float v0 = h.x * sc.x + bi.x;
    float v1 = h.y * sc.y + bi.y;
    float v2 = h.z * sc.z + bi.z;
    float v3 = h.w * sc.w + bi.w;
    if (resid) {
        float4 r = *reinterpret_cast<const float4*>(resid + idx);
        v0 += r.x; v1 += r.y; v2 += r.z; v3 += r.w;
    }
    if (do_relu) {
        v0 = fmaxf(v0, 0.f); v1 = fmaxf(v1, 0.f);
        v2 = fmaxf(v2, 0.f); v3 = fmaxf(v3, 0.f);
    }
    if (out_f32) {
        float4 o; o.x = v0; o.y = v1; o.z = v2; o.w = v3;
        *reinterpret_cast<float4*>(out_f32 + idx) = o;
    }
    union { bf16 b[4]; uint2 u; } pk;
    pk.b[0] = (bf16)v0; pk.b[1] = (bf16)v1; pk.b[2] = (bf16)v2; pk.b[3] = (bf16)v3;
    *reinterpret_cast<uint2*>(out_bf + idx) = pk.u;
}

// ---------------------------------------------------------------------------
// logits = x4 @ w_out^T + b_out  (128 -> 8, fp32 out)
// ---------------------------------------------------------------------------
__device__ inline float bf_lo(unsigned u) { union { unsigned x; float f; } t; t.x = u << 16;         return t.f; }
__device__ inline float bf_hi(unsigned u) { union { unsigned x; float f; } t; t.x = u & 0xffff0000u; return t.f; }

__global__ __launch_bounds__(256) void logits_kernel(
    const bf16* __restrict__ X, const float* __restrict__ w_out,
    const float* __restrict__ b_out, float* __restrict__ out, int nrows) {
    __shared__ float sw[8 * 128];
    __shared__ float sb[8];
    for (int i = threadIdx.x; i < 1024; i += 256) sw[i] = w_out[i];
    if (threadIdx.x < 8) sb[threadIdx.x] = b_out[threadIdx.x];
    __syncthreads();
    int row = blockIdx.x * blockDim.x + threadIdx.x;
    if (row >= nrows) return;
    const bf16* xr = X + (size_t)row * 128;
    float acc[8] = {0.f, 0.f, 0.f, 0.f, 0.f, 0.f, 0.f, 0.f};
#pragma unroll
    for (int k = 0; k < 128; k += 8) {
        uint4 u = *reinterpret_cast<const uint4*>(xr + k);
        float xv[8] = { bf_lo(u.x), bf_hi(u.x), bf_lo(u.y), bf_hi(u.y),
                        bf_lo(u.z), bf_hi(u.z), bf_lo(u.w), bf_hi(u.w) };
#pragma unroll
        for (int j = 0; j < 8; ++j) {
            const float* wj = sw + j * 128 + k;
#pragma unroll
            for (int e = 0; e < 8; ++e) acc[j] += xv[e] * wj[e];
        }
    }
    float4 o0 = make_float4(acc[0] + sb[0], acc[1] + sb[1], acc[2] + sb[2], acc[3] + sb[3]);
    float4 o1 = make_float4(acc[4] + sb[4], acc[5] + sb[5], acc[6] + sb[6], acc[7] + sb[7]);
    *reinterpret_cast<float4*>(out + (size_t)row * 8)     = o0;
    *reinterpret_cast<float4*>(out + (size_t)row * 8 + 4) = o1;
}

__global__ void f32_to_bf16_kernel(const float* __restrict__ in,
                                   bf16* __restrict__ out, int n) {
    int i = blockIdx.x * blockDim.x + threadIdx.x;
    if (i < n) out[i] = (bf16)in[i];
}

// ---------------------------------------------------------------------------
// Host side
// ---------------------------------------------------------------------------
extern "C" void kernel_launch(void* const* d_in, const int* in_sizes, int n_in,
                              void* d_out, int out_size, void* d_ws, size_t ws_size,
                              hipStream_t stream) {
    (void)in_sizes; (void)n_in; (void)out_size; (void)ws_size;
    const int N = 120000;

    const float* feat  = (const float*)d_in[0];
    // d_in[1]=coord, d_in[2]=knn_idx, d_in[3]=offset: dead code in reference
    const float* w1    = (const float*)d_in[4];
    const float* g1    = (const float*)d_in[5];
    const float* b1    = (const float*)d_in[6];
    const float* w2    = (const float*)d_in[7];
    const float* g2    = (const float*)d_in[8];
    const float* b2    = (const float*)d_in[9];
    const float* w3    = (const float*)d_in[10];
    const float* g3    = (const float*)d_in[11];
    const float* b3    = (const float*)d_in[12];
    const float* w4    = (const float*)d_in[13];
    const float* g4    = (const float*)d_in[14];
    const float* b4    = (const float*)d_in[15];
    const float* w_out = (const float*)d_in[16];
    const float* b_out = (const float*)d_in[17];
    float* logits = (float*)d_out;

    // Workspace layout (all 16B aligned)
    char* ws = (char*)d_ws;
    bf16*  Bbf = (bf16*)ws;                                   // N*256 bf16
    float* B1  = (float*)(ws + (size_t)N * 256 * 2);          // N*256 f32 (pre-BN H)
    float* B2  = (float*)(ws + (size_t)N * 256 * 2
                             + (size_t)N * 256 * 4);          // N*128 f32 (identity)
    bf16*  Wbf = (bf16*)(ws + (size_t)N * 256 * 2
                            + (size_t)N * 256 * 4
                            + (size_t)N * 128 * 4);           // bf16 weights
    float* stats = (float*)((char*)Wbf + (16384 + 32768 + 32768) * 2);
    float* s_sum = stats, * s_sq = stats + 256, * s_scale = stats + 512, * s_bias = stats + 768;

    bf16* w2bf = Wbf;
    bf16* w3bf = Wbf + 16384;
    bf16* w4bf = Wbf + 49152;

    // Convert weights to bf16 (L2-resident for the GEMM sweeps)
    f32_to_bf16_kernel<<<64, 256, 0, stream>>>(w2, w2bf, 16384);
    f32_to_bf16_kernel<<<128, 256, 0, stream>>>(w3, w3bf, 32768);
    f32_to_bf16_kernel<<<128, 256, 0, stream>>>(w4, w4bf, 32768);

    auto finalize = [&](int width, const float* gamma, const float* beta) {
        finalize_kernel<<<1, 256, 0, stream>>>(s_sum, s_sq, gamma, beta,
                                               s_scale, s_bias, width, 1.0f / N);
    };
    auto norm = [&](int width, const float* resid, float* out_f32, int relu) {
        int total  = N * width;
        int blocks = (total / 4 + 255) / 256;
        normalize_kernel<<<blocks, 256, 0, stream>>>(B1, s_scale, s_bias, resid,
                                                     out_f32, Bbf, total, width, relu);
    };

    const int GEMM_BLOCKS = (N / 16) / 4;  // 1875: 4 waves/block, 16 rows/wave

    // L1: H = feat @ w1^T (stats fused) ; x1 = relu(BN(H)) -> Bbf
    zero_stats_kernel<<<1, 256, 0, stream>>>(s_sum, s_sq, 128);
    layer1_kernel<<<512, 256, 0, stream>>>(feat, w1, B1, s_sum, s_sq, N);
    finalize(128, g1, b1);
    norm(128, nullptr, nullptr, 1);

    // (KNN gather + max is dead code: concat()[:, :128] == x1)

    // L2: H = x1 @ w2^T (stats fused) ; x2 = relu(BN(H)) -> Bbf, identity -> B2
    zero_stats_kernel<<<1, 256, 0, stream>>>(s_sum, s_sq, 128);
    gemm_bf16_kernel<4, 8><<<GEMM_BLOCKS, 128, 0, stream>>>(Bbf, w2bf, B1, s_sum, s_sq, N);
    finalize(128, g2, b2);
    norm(128, nullptr, B2, 1);

    // L3: H = x2 @ w3^T (stats fused) ; x3 = relu(BN(H)) -> Bbf (width 256)
    zero_stats_kernel<<<1, 256, 0, stream>>>(s_sum, s_sq, 256);
    gemm_bf16_kernel<4, 16><<<GEMM_BLOCKS, 128, 0, stream>>>(Bbf, w3bf, B1, s_sum, s_sq, N);
    finalize(256, g3, b3);
    norm(256, nullptr, nullptr, 1);

    // L4: H = x3 @ w4^T (stats fused) ; x4 = relu(BN(H) + identity) -> Bbf
    zero_stats_kernel<<<1, 256, 0, stream>>>(s_sum, s_sq, 128);
    gemm_bf16_kernel<8, 8><<<GEMM_BLOCKS, 128, 0, stream>>>(Bbf, w4bf, B1, s_sum, s_sq, N);
    finalize(128, g4, b4);
    norm(128, B2, nullptr, 1);

    // L5: logits = x4 @ w_out^T + b_out
    logits_kernel<<<(N + 255) / 256, 256, 0, stream>>>(Bbf, w_out, b_out, logits, N);
}
